// HM_LSTMCell_45878840656512
// MI455X (gfx1250) — matrix operand order repaired and checked by reading, not compiled, not run on hardware
//
#include <hip/hip_runtime.h>
#include <hip/hip_bf16.h>

#define BATCH   16384
#define HIDN    512
#define KTOT    1536          // 3 * 512 stacked K
#define NCOL    2049          // 4*HIDN + 1
#define BM      64            // rows per block
#define ELP     1544          // padded LDS row stride in bf16 (3088 B, 16B aligned)
#define NKT     (KTOT / 16)   // 96 K-blocks of 16
// packed-weight geometry: block (ct, kt) holds 16 cols x 16 k, lane-major:
//   Wp[(((ct*NKT + kt)*16 + (c&15))*16 + (k&15)]
#define GATE_OFF ((size_t)32 * NKT * 256)   // elems between gates (32 col-tiles)

typedef __bf16 bf16;
typedef __attribute__((ext_vector_type(16))) __bf16 v16bf;
typedef __attribute__((ext_vector_type(8)))  __bf16 v8bf;
typedef __attribute__((ext_vector_type(4)))  __bf16 v4bf;
typedef __attribute__((ext_vector_type(8)))  float   v8f;
typedef __attribute__((ext_vector_type(4)))  float   v4f;

__device__ __forceinline__ float fsigmoid(float x) {
    return 1.0f / (1.0f + __expf(-x));
}
__device__ __forceinline__ float ftanh_fast(float x) {
    return 1.0f - 2.0f / (__expf(2.0f * x) + 1.0f);
}

// ---------------------------------------------------------------------------
// Kernel 1: transpose + bf16-convert stacked weights [1536 x 2049] into a
// fragment-major packed layout: per (col-tile ct, k-tile kt) a 16x16 bf16
// block with each column's 16 k-values contiguous (32 B per lane). WMMA
// B-fragment loads become fully coalesced (512 B per half-wave).
// Column 2048 (boundary detector) kept in f32 as wlast[k].
// ---------------------------------------------------------------------------
__global__ void __launch_bounds__(256)
prep_weights(const float* __restrict__ U11, const float* __restrict__ U21,
             const float* __restrict__ W01, bf16* __restrict__ Wp,
             float* __restrict__ wlast)
{
    __shared__ float tile[32][33];
    const int k0 = blockIdx.x * 32;
    const int c0 = blockIdx.y * 32;
    const int tx = threadIdx.x;     // 0..31
    const int ty = threadIdx.y;     // 0..7

    #pragma unroll
    for (int j = 0; j < 4; ++j) {
        int k = k0 + ty + j * 8;          // k < 1536 always (grid exact)
        int c = c0 + tx;
        float v = 0.0f;
        if (c < NCOL) {
            const float* src; int kr;
            if (k < 512)       { src = U11; kr = k;        }
            else if (k < 1024) { src = U21; kr = k - 512;  }
            else               { src = W01; kr = k - 1024; }
            v = src[(size_t)kr * NCOL + c];       // coalesced in c
        }
        tile[ty + j * 8][tx] = v;
    }
    __syncthreads();
    #pragma unroll
    for (int j = 0; j < 4; ++j) {
        int c = c0 + ty + j * 8;
        int k = k0 + tx;
        float v = tile[tx][ty + j * 8];
        if (c < 4 * HIDN) {
            size_t idx = (((size_t)(c >> 4) * NKT + (k >> 4)) * 16 + (c & 15)) * 16
                         + (k & 15);
            Wp[idx] = (bf16)v;                    // coalesced: k&15 fastest
        } else if (c == 4 * HIDN) {
            wlast[k] = v;
        }
    }
}

// ---------------------------------------------------------------------------
// Kernel 2: fused HM-LSTM cell. One block = 64 batch rows.
//   - stage A' = [h | z*h_top | zb*h_bottom] as bf16 into LDS (64 x 1544 pad)
//   - 8 waves: wave = (mtile in [0,4), n-half); per n-tile keep 4 gate
//     accumulators and run K=1536 with v_wmma_f32_16x16x32_bf16
//   - in-register sigmoid/tanh epilogue -> h_new, c_new
//   - z_hat column: per-row 1536-dot against wlast from the LDS tile
// ---------------------------------------------------------------------------
__global__ void __launch_bounds__(256)
hmlstm_main(const float* __restrict__ cIn,  const float* __restrict__ hBot,
            const float* __restrict__ hIn,  const float* __restrict__ hTop,
            const float* __restrict__ Z,    const float* __restrict__ ZB,
            const bf16*  __restrict__ Wp,   const float* __restrict__ wlast,
            const float* __restrict__ bias, float* __restrict__ out)
{
    extern __shared__ char smemRaw[];
    bf16*  ldsA  = (bf16*)smemRaw;                         // BM x ELP bf16
    float* ldsZ  = (float*)(smemRaw + (size_t)BM * ELP * 2);
    float* ldsZB = ldsZ + BM;

    const int tid  = threadIdx.x;
    const int row0 = blockIdx.x * BM;

    if (tid < BM) {
        ldsZ[tid]  = Z[row0 + tid];
        ldsZB[tid] = ZB[row0 + tid];
    }

    // ---- stage scaled bf16 A' tile into LDS (96 iters of float4) ----------
    #pragma unroll 4
    for (int it = 0; it < (BM * KTOT / 4) / 256; ++it) {
        int idx = it * 256 + tid;               // group-of-4 index
        int m   = idx / (KTOT / 4);
        int q   = idx - m * (KTOT / 4);
        int k   = q * 4;
        int seg = k >> 9;                       // 0:h  1:h_top  2:h_bottom
        int ks  = k & 511;
        int row = row0 + m;
        const float* src = (seg == 0) ? hIn : (seg == 1 ? hTop : hBot);
        float scale = (seg == 0) ? 1.0f : (seg == 1 ? Z[row] : ZB[row]);
        v4f f = *(const v4f*)(src + (size_t)row * HIDN + ks);
        v4bf o;
        o.x = (bf16)(f.x * scale); o.y = (bf16)(f.y * scale);
        o.z = (bf16)(f.z * scale); o.w = (bf16)(f.w * scale);
        *(v4bf*)(ldsA + (size_t)m * ELP + k) = o;
    }
    __syncthreads();

    const int wave  = tid >> 5;
    const int lane  = tid & 31;
    const int mtile = wave >> 1;      // which 16-row slab of the 64
    const int halfN = wave & 1;       // which 16 n-tiles of the 32
    const int lh    = lane >> 4;      // lane half (0/1)
    const int ln    = lane & 15;

    float* hOut = out;
    float* cOut = out + (size_t)BATCH * HIDN;
    float* zOut = out + (size_t)2 * BATCH * HIDN;

    // A fragment: lane row = ln, K chunk [kk+16*lh, +16) contiguous
    const bf16* aP = ldsA + (size_t)(mtile * 16 + ln) * ELP + 16 * lh;

    for (int nt = halfN * 16; nt < halfN * 16 + 16; ++nt) {
        const int jj = nt * 16 + ln;          // output column 0..511
        v8f accF = {}; v8f accI = {}; v8f accO = {}; v8f accG = {};
        // packed B base for gate 0, col-tile nt: lane chunk (kt = kk/16 + lh)
        // lives at bB + kk*16 (+8 for second v8bf); gate g adds g*GATE_OFF.
        const bf16* bB = Wp + (((size_t)nt * NKT + lh) * 16 + ln) * 16;

        #pragma unroll 2
        for (int kk = 0; kk < KTOT; kk += 32) {
            v8bf a0 = *(const v8bf*)(aP + kk);
            v8bf a1 = *(const v8bf*)(aP + kk + 8);
            v16bf Af = __builtin_shufflevector(a0, a1,
                        0,1,2,3,4,5,6,7,8,9,10,11,12,13,14,15);
            const bf16* bK = bB + (size_t)kk * 16;
            {
                v8bf b0 = *(const v8bf*)(bK);
                v8bf b1 = *(const v8bf*)(bK + 8);
                v16bf Bf = __builtin_shufflevector(b0, b1,
                            0,1,2,3,4,5,6,7,8,9,10,11,12,13,14,15);
                accF = __builtin_amdgcn_wmma_f32_16x16x32_bf16(
                           false, Af, false, Bf, (short)0, accF, false, false);
            }
            {
                v8bf b0 = *(const v8bf*)(bK + GATE_OFF);
                v8bf b1 = *(const v8bf*)(bK + GATE_OFF + 8);
                v16bf Bf = __builtin_shufflevector(b0, b1,
                            0,1,2,3,4,5,6,7,8,9,10,11,12,13,14,15);
                accI = __builtin_amdgcn_wmma_f32_16x16x32_bf16(
                           false, Af, false, Bf, (short)0, accI, false, false);
            }
            {
                v8bf b0 = *(const v8bf*)(bK + 2 * GATE_OFF);
                v8bf b1 = *(const v8bf*)(bK + 2 * GATE_OFF + 8);
                v16bf Bf = __builtin_shufflevector(b0, b1,
                            0,1,2,3,4,5,6,7,8,9,10,11,12,13,14,15);
                accO = __builtin_amdgcn_wmma_f32_16x16x32_bf16(
                           false, Af, false, Bf, (short)0, accO, false, false);
            }
            {
                v8bf b0 = *(const v8bf*)(bK + 3 * GATE_OFF);
                v8bf b1 = *(const v8bf*)(bK + 3 * GATE_OFF + 8);
                v16bf Bf = __builtin_shufflevector(b0, b1,
                            0,1,2,3,4,5,6,7,8,9,10,11,12,13,14,15);
                accG = __builtin_amdgcn_wmma_f32_16x16x32_bf16(
                           false, Af, false, Bf, (short)0, accG, false, false);
            }
        }

        // ---- fused LSTM epilogue for this 16x16 tile -----------------------
        const float biasF = bias[jj];
        const float biasI = bias[HIDN + jj];
        const float biasO = bias[2 * HIDN + jj];
        const float biasG = bias[3 * HIDN + jj];
        #pragma unroll
        for (int e = 0; e < 8; ++e) {
            // C/D layout: VGPR e holds local rows e (lanes 0-15) / e+8 (16-31)
            int lrow = mtile * 16 + e + 8 * lh;
            int row  = row0 + lrow;
            float zr  = ldsZ[lrow];
            float zbr = ldsZB[lrow];
            size_t off = (size_t)row * HIDN + jj;
            float c_old = cIn[off];
            float h_old = hIn[off];
            float f = fsigmoid(accF[e] + biasF);
            float i = fsigmoid(accI[e] + biasI);
            float o = fsigmoid(accO[e] + biasO);
            float g = ftanh_fast(accG[e] + biasG);
            float ig   = i * g;
            float omz  = 1.0f - zr;
            float omzb = 1.0f - zbr;
            float cn = zr * ig + omz * omzb * c_old + omz * zbr * (f * c_old + ig);
            float hn = omz * omzb * h_old + (zr + omz * zbr) * o * ftanh_fast(cn);
            hOut[off] = hn;
            cOut[off] = cn;
        }
    }

    // ---- z_hat column (col 2048): 1536-dot per row from the LDS tile ------
    if (tid < BM) {
        const bf16* r = ldsA + (size_t)tid * ELP;
        float acc = 0.0f;
        for (int k = 0; k < KTOT; k += 8) {
            v8bf v = *(const v8bf*)(r + k);
            #pragma unroll
            for (int t = 0; t < 8; ++t)
                acc += (float)v[t] * wlast[k + t];
        }
        acc += bias[4 * HIDN];
        float zh = (acc * 1.0f + 1.0f) * 0.5f;          // A (slope) = 1.0
        zh = fminf(fmaxf(zh, 0.0f), 1.0f);
        // straight-through forward value: round-half-even like jnp.round
        zOut[row0 + tid] = rintf(zh);
    }
}

// ---------------------------------------------------------------------------
extern "C" void kernel_launch(void* const* d_in, const int* in_sizes, int n_in,
                              void* d_out, int out_size, void* d_ws, size_t ws_size,
                              hipStream_t stream) {
    (void)in_sizes; (void)n_in; (void)out_size; (void)ws_size;
    const float* cIn  = (const float*)d_in[0];
    const float* hBot = (const float*)d_in[1];
    const float* hIn  = (const float*)d_in[2];
    const float* hTop = (const float*)d_in[3];
    const float* Z    = (const float*)d_in[4];
    const float* ZB   = (const float*)d_in[5];
    const float* U11  = (const float*)d_in[6];
    const float* U21  = (const float*)d_in[7];
    const float* W01  = (const float*)d_in[8];
    const float* bias = (const float*)d_in[9];

    bf16*  Wp    = (bf16*)d_ws;
    float* wlast = (float*)((char*)d_ws + (size_t)4 * HIDN * KTOT * sizeof(bf16));

    // weight transpose + fragment-pack (L2-resident bf16 weights, 6.3 MB)
    dim3 tb(32, 8);
    dim3 tg(KTOT / 32, (NCOL + 31) / 32);
    prep_weights<<<tg, tb, 0, stream>>>(U11, U21, W01, Wp, wlast);

    size_t lds = (size_t)BM * ELP * 2 + (size_t)2 * BM * sizeof(float);
    (void)hipFuncSetAttribute((const void*)hmlstm_main,
                              hipFuncAttributeMaxDynamicSharedMemorySize,
                              (int)lds);
    hmlstm_main<<<BATCH / BM, 256, lds, stream>>>(
        cIn, hBot, hIn, hTop, Z, ZB, Wp, wlast, bias, (float*)d_out);
}
